// ParallelSelfAttention_11115375362456
// MI455X (gfx1250) — matrix operand order repaired and checked by reading, not compiled, not run on hardware
//
#include <hip/hip_runtime.h>

// ---------------- problem constants ----------------
#define Bb 2
#define Ss 2048
#define Hh 2048
#define NHEAD 16
#define HN 128
#define H3 (3*Hh)

typedef __bf16 bf16;
typedef __attribute__((ext_vector_type(16))) __bf16 v16bf;
typedef __attribute__((ext_vector_type(8)))  __bf16 v8bf;
typedef __attribute__((ext_vector_type(8)))  float  v8f;

// ---------------- WMMA helpers ----------------
static __device__ inline v16bf cat16(v8bf lo, v8bf hi) {
  return __builtin_shufflevector(lo, hi, 0,1,2,3,4,5,6,7,8,9,10,11,12,13,14,15);
}

// A-operand (16x32 bf16, M x K) from a row-major tile.
// Lane (hf, m): elems 0..7 -> K = kofs+8*hf+e ; elems 8..15 -> K = kofs+16+8*hf+(e-8)
static __device__ inline v16bf load_a(const bf16* base, int ld, int m, int kofs, int hf) {
  const bf16* p = base + m*ld + kofs + 8*hf;
  v8bf lo = *(const v8bf*)p;
  v8bf hi = *(const v8bf*)(p + 16);
  return cat16(lo, hi);
}
// B-operand (32x16 bf16, K x N) where B column n is a contiguous memory row.
// Lane (hf, n): elems 0..7 -> K = kofs+8*hf+e ; elems 8..15 -> K = kofs+16+8*hf+(e-8)
static __device__ inline v16bf load_b(const bf16* base, int ld, int n, int kofs, int hf) {
  const bf16* p = base + n*ld + kofs + 8*hf;
  v8bf lo = *(const v8bf*)p;
  v8bf hi = *(const v8bf*)(p + 16);
  return cat16(lo, hi);
}
static __device__ inline v8f wmma_bf16(v16bf a, v16bf b, v8f c) {
  return __builtin_amdgcn_wmma_f32_16x16x32_bf16(false, a, false, b, (short)0, c, false, false);
}

// CDNA5 async global->LDS copy (16B), tracked by ASYNCcnt.
static __device__ inline void async_b128(void* lds_ptr, const void* gptr) {
  unsigned int loff = (unsigned int)(unsigned long long)lds_ptr;
  asm volatile("global_load_async_to_lds_b128 %0, %1, off"
               :: "v"(loff), "v"(gptr) : "memory");
}
static __device__ inline void async_wait0() {
  asm volatile("s_wait_asynccnt 0" ::: "memory");
}

// ---------------- converts ----------------
__global__ void k_cvt(const float* __restrict__ in, bf16* __restrict__ out, int n) {
  int i = blockIdx.x * blockDim.x + threadIdx.x;
  if (i < n) out[i] = (bf16)in[i];
}
// in: K x N row-major f32 -> out: N x K row-major bf16 (transposed)
__global__ void k_cvt_t(const float* __restrict__ in, bf16* __restrict__ out, int K, int N) {
  int i = blockIdx.x * blockDim.x + threadIdx.x;
  if (i < K * N) {
    int k = i / N, n = i - k * N;
    out[(size_t)n * K + k] = (bf16)in[i];
  }
}

// ---------------- generic bf16 WMMA GEMM ----------------
// C[M,N] = A[M,K] @ Bt[N,K]^T + bias[N];  M,N multiples of 64, K multiple of 32.
template<bool OUT_BF16>
__global__ __launch_bounds__(128) void k_gemm(const bf16* __restrict__ A,
                                              const bf16* __restrict__ Bt,
                                              const float* __restrict__ bias,
                                              void* __restrict__ Cout,
                                              int M, int N, int K) {
  __shared__ bf16 at[64 * 40];
  __shared__ bf16 bt[64 * 40];
  const int tid = threadIdx.x;
  const int wv = tid >> 5, lane = tid & 31, hf = lane >> 4, hm = lane & 15;
  const int m0 = blockIdx.y * 64, n0 = blockIdx.x * 64;
  v8f acc[4] = {};
  for (int ko = 0; ko < K; ko += 32) {
    __syncthreads();
    {
      int r = tid >> 1, seg = tid & 1;
      const bf16* sa = A  + (size_t)(m0 + r) * K + ko + seg * 16;
      const bf16* sb = Bt + (size_t)(n0 + r) * K + ko + seg * 16;
      *(v8bf*)(at + r * 40 + seg * 16)     = *(const v8bf*)(sa);
      *(v8bf*)(at + r * 40 + seg * 16 + 8) = *(const v8bf*)(sa + 8);
      *(v8bf*)(bt + r * 40 + seg * 16)     = *(const v8bf*)(sb);
      *(v8bf*)(bt + r * 40 + seg * 16 + 8) = *(const v8bf*)(sb + 8);
    }
    __syncthreads();
    v16bf a = load_a(at + wv * 16 * 40, 40, hm, 0, hf);
#pragma unroll
    for (int ct = 0; ct < 4; ++ct) {
      v16bf b = load_b(bt, 40, ct * 16 + hm, 0, hf);
      acc[ct] = wmma_bf16(a, b, acc[ct]);
    }
  }
#pragma unroll
  for (int ct = 0; ct < 4; ++ct) {
    float bv = bias[n0 + ct * 16 + hm];
#pragma unroll
    for (int r = 0; r < 8; ++r) {
      int m = wv * 16 + r + 8 * hf;
      float v = acc[ct][r] + bv;
      size_t off = (size_t)(m0 + m) * N + n0 + ct * 16 + hm;
      if (OUT_BF16) ((bf16*)Cout)[off] = (bf16)v;
      else          ((float*)Cout)[off] = v;
    }
  }
}

// ---------------- fused rel-attention ----------------
// grid: (S/64, NHEAD, B); 128 threads = 4 waves; wave w owns 16 query rows.
__global__ __launch_bounds__(128) void k_attn(const bf16* __restrict__ qkv, // (B*S) x 3H
                                              const bf16* __restrict__ rel, // (B*S) x H
                                              const float* __restrict__ rwb, // N x HN
                                              const float* __restrict__ rrb, // N x HN
                                              bf16* __restrict__ ctx) {      // (B*S) x H
  __shared__ bf16  kq[64 * 136];     // K tile (also raw-Q staging); padded stride
  __shared__ bf16  qws[64 * 136];    // q + r_w_bias (bf16, row-major)
  __shared__ bf16  qrs[64 * 136];    // q + r_r_bias
  __shared__ bf16  vt[128 * 72];     // V tile transposed [d][j]
  __shared__ bf16  rl[128 * 136];    // rel slice (128 rows)
  __shared__ float tb[4][16 * 84];   // per-wave bd staging for the shift
  __shared__ bf16  pb[4][16 * 72];   // per-wave probs staging

  const int tid = threadIdx.x;
  const int wv = tid >> 5, lane = tid & 31, hf = lane >> 4, hm = lane & 15;
  const int i0 = blockIdx.x * 64;
  const int nh = blockIdx.y;
  const int b  = blockIdx.z;
  const int i0w = i0 + wv * 16;

  const size_t rq = (size_t)(b * Ss) * H3;
  const bf16* qbase = qkv + rq + nh * HN;
  const bf16* kbase = qkv + rq + Hh + nh * HN;
  const bf16* vbase = qkv + rq + 2 * Hh + nh * HN;
  const bf16* rbase = rel + (size_t)(b * Ss) * Hh + nh * HN;

  // stage raw Q block (64 x 128) into kq (async copy)
  for (int c = tid; c < 64 * 16; c += 128) {
    int r = c >> 4, c8 = c & 15;
    async_b128(kq + r * 136 + c8 * 8, qbase + (size_t)(i0 + r) * H3 + c8 * 8);
  }
  async_wait0();
  __syncthreads();

  // build q+rwb and q+rrb in LDS (row-major, consumed as WMMA A-operands later)
  for (int c = tid; c < 64 * 16; c += 128) {
    int r = c >> 4, c8 = c & 15;
    v8bf q = *(const v8bf*)(kq + r * 136 + c8 * 8);
    const float* bw = rwb + nh * HN + c8 * 8;
    const float* br = rrb + nh * HN + c8 * 8;
    v8bf aw, ar;
#pragma unroll
    for (int e = 0; e < 8; ++e) {
      float qf = (float)q[e];
      aw[e] = (bf16)(qf + bw[e]);
      ar[e] = (bf16)(qf + br[e]);
    }
    *(v8bf*)(qws + r * 136 + c8 * 8) = aw;
    *(v8bf*)(qrs + r * 136 + c8 * 8) = ar;
  }
  __syncthreads();

  const bf16* qwb = qws + wv * 16 * 136;   // this wave's 16 query rows
  const bf16* qrb = qrs + wv * 16 * 136;

  v8f o[8] = {};
  float mst[8], lst[8];
#pragma unroll
  for (int r = 0; r < 8; ++r) { mst[r] = -3.0e38f; lst[r] = 0.0f; }

  const float scale = 0.08838834764831845f;   // 1/sqrt(128)
  const int pbase = Ss - 64 - i0;             // rel slice base (+ j0 per block)

  for (int j0 = 0; j0 <= i0; j0 += 64) {
    __syncthreads();   // previous iteration consumers done
    // K tile (async)
    for (int c = tid; c < 64 * 16; c += 128) {
      int r = c >> 4, c8 = c & 15;
      async_b128(kq + r * 136 + c8 * 8, kbase + (size_t)(j0 + r) * H3 + c8 * 8);
    }
    // rel slice: rows pbase+j0 .. +127 (clamped; clamped rows feed masked scores only)
    for (int c = tid; c < 128 * 16; c += 128) {
      int r = c >> 4, c8 = c & 15;
      int p = pbase + j0 + r; if (p > Ss - 1) p = Ss - 1;
      async_b128(rl + r * 136 + c8 * 8, rbase + (size_t)p * Hh + c8 * 8);
    }
    // V tile, transposed into vt[d][j] (through VGPRs)
    for (int c = tid; c < 64 * 16; c += 128) {
      int r = c >> 4, c8 = c & 15;
      v8bf vv = *(const v8bf*)(vbase + (size_t)(j0 + r) * H3 + c8 * 8);
#pragma unroll
      for (int e = 0; e < 8; ++e) vt[(c8 * 8 + e) * 72 + r] = vv[e];
    }
    async_wait0();
    __syncthreads();

    // ---- content scores ac = (q+rwb) @ K^T : ct-outer, ks-inner ----
    v8f ac[4];
#pragma unroll
    for (int ct = 0; ct < 4; ++ct) {
      v8f a = {};
#pragma unroll
      for (int ks = 0; ks < 4; ++ks) {
        v16bf qa = load_a(qwb, 136, hm, 32 * ks, hf);
        v16bf bb = load_b(kq, 136, ct * 16 + hm, 32 * ks, hf);
        a = wmma_bf16(qa, bb, a);
      }
      ac[ct] = a;
    }

    // ---- position scores T = (q+rrb) @ rel_slice^T, spilled tile-by-tile ----
    float* tw = tb[wv];
#pragma unroll
    for (int pt = 0; pt < 5; ++pt) {
      v8f t = {};
#pragma unroll
      for (int ks = 0; ks < 4; ++ks) {
        v16bf qa = load_a(qrb, 136, hm, 32 * ks, hf);
        v16bf bb = load_b(rl, 136, (48 - 16 * wv) + pt * 16 + hm, 32 * ks, hf);
        t = wmma_bf16(qa, bb, t);
      }
#pragma unroll
      for (int r = 0; r < 8; ++r)
        tw[(r + 8 * hf) * 84 + pt * 16 + hm] = t[r];
    }

    // ---- shift + mask + online softmax (probs exp'ed in place in ac) ----
    float rmax[8];
#pragma unroll
    for (int r = 0; r < 8; ++r) rmax[r] = -3.0e38f;
#pragma unroll
    for (int ct = 0; ct < 4; ++ct) {
#pragma unroll
      for (int r = 0; r < 8; ++r) {
        int m = r + 8 * hf;
        float bd = tw[m * 84 + (15 - m + ct * 16 + hm)];
        float sc = (ac[ct][r] + bd) * scale;
        int ig = i0w + m, jg = j0 + ct * 16 + hm;
        sc = (jg <= ig) ? sc : -65504.0f;
        ac[ct][r] = sc;
        rmax[r] = fmaxf(rmax[r], sc);
      }
    }
#pragma unroll
    for (int r = 0; r < 8; ++r) {
      float v = rmax[r];
#pragma unroll
      for (int s = 1; s < 16; s <<= 1) v = fmaxf(v, __shfl_xor(v, s, 32));
      rmax[r] = v;
    }
    float fsc[8];
#pragma unroll
    for (int r = 0; r < 8; ++r) {
      float mn = fmaxf(mst[r], rmax[r]);
      fsc[r] = __expf(mst[r] - mn);
      mst[r] = mn;
    }
    float rsum[8];
#pragma unroll
    for (int r = 0; r < 8; ++r) rsum[r] = 0.0f;
#pragma unroll
    for (int ct = 0; ct < 4; ++ct)
#pragma unroll
      for (int r = 0; r < 8; ++r) {
        float e = __expf(ac[ct][r] - mst[r]);
        ac[ct][r] = e;
        rsum[r] += e;
      }
#pragma unroll
    for (int r = 0; r < 8; ++r) {
      float v = rsum[r];
#pragma unroll
      for (int s = 1; s < 16; s <<= 1) v += __shfl_xor(v, s, 32);
      lst[r] = lst[r] * fsc[r] + v;
    }
#pragma unroll
    for (int dt = 0; dt < 8; ++dt)
#pragma unroll
      for (int r = 0; r < 8; ++r) o[dt][r] *= fsc[r];

    // ---- stage probs, then P @ V via WMMA ----
    bf16* pw = pb[wv];
#pragma unroll
    for (int ct = 0; ct < 4; ++ct)
#pragma unroll
      for (int r = 0; r < 8; ++r)
        pw[(r + 8 * hf) * 72 + ct * 16 + hm] = (bf16)ac[ct][r];

#pragma unroll
    for (int kst = 0; kst < 2; ++kst) {
      v16bf pa = load_a(pw, 72, hm, 32 * kst, hf);
#pragma unroll
      for (int dt = 0; dt < 8; ++dt) {
        v16bf vb = load_b(vt, 72, dt * 16 + hm, 32 * kst, hf);
        o[dt] = wmma_bf16(pa, vb, o[dt]);
      }
    }
  }

  // finalize: divide by row sums, store ctx (bf16) in [B*S, H] with head offset
#pragma unroll
  for (int r = 0; r < 8; ++r) {
    float inv = 1.0f / lst[r];
    int m = r + 8 * hf;
    size_t orow = (size_t)(b * Ss + i0w + m) * Hh + nh * HN;
#pragma unroll
    for (int dt = 0; dt < 8; ++dt)
      ctx[orow + dt * 16 + hm] = (bf16)(o[dt][r] * inv);
  }
}

// ---------------- host launcher ----------------
extern "C" void kernel_launch(void* const* d_in, const int* in_sizes, int n_in,
                              void* d_out, int out_size, void* d_ws, size_t ws_size,
                              hipStream_t stream) {
  (void)in_sizes; (void)n_in; (void)out_size; (void)ws_size;
  const float* hs   = (const float*)d_in[0];
  const float* pe   = (const float*)d_in[1];
  // d_in[2] = ltor_mask (causal handled analytically)
  const float* Wqkv = (const float*)d_in[3];
  const float* bqkv = (const float*)d_in[4];
  const float* Wrel = (const float*)d_in[5];
  const float* brel = (const float*)d_in[6];
  const float* Wdns = (const float*)d_in[7];
  const float* bdns = (const float*)d_in[8];
  const float* rwb  = (const float*)d_in[9];
  const float* rrb  = (const float*)d_in[10];

  char* ws = (char*)d_ws;
  size_t off = 0;
  auto alloc = [&](size_t bytes) {
    void* p = ws + off;
    off += (bytes + 255) & ~(size_t)255;
    return p;
  };
  const int MBS = Bb * Ss;                        // 4096
  bf16* Xh   = (bf16*)alloc((size_t)MBS * Hh * 2);
  bf16* Xp   = (bf16*)alloc((size_t)MBS * Hh * 2);
  bf16* WqkT = (bf16*)alloc((size_t)H3 * Hh * 2);
  bf16* WreT = (bf16*)alloc((size_t)Hh * Hh * 2);
  bf16* WdnT = (bf16*)alloc((size_t)Hh * Hh * 2);
  bf16* QKV  = (bf16*)alloc((size_t)MBS * H3 * 2);
  bf16* REL  = (bf16*)alloc((size_t)MBS * Hh * 2);
  bf16* CTX  = (bf16*)alloc((size_t)MBS * Hh * 2);

  const int nX = MBS * Hh;
  k_cvt<<<(nX + 255) / 256, 256, 0, stream>>>(hs, Xh, nX);
  k_cvt<<<(nX + 255) / 256, 256, 0, stream>>>(pe, Xp, nX);
  k_cvt_t<<<(Hh * H3 + 255) / 256, 256, 0, stream>>>(Wqkv, WqkT, Hh, H3);
  k_cvt_t<<<(Hh * Hh + 255) / 256, 256, 0, stream>>>(Wrel, WreT, Hh, Hh);
  k_cvt_t<<<(Hh * Hh + 255) / 256, 256, 0, stream>>>(Wdns, WdnT, Hh, Hh);

  dim3 blk(128);
  k_gemm<true><<<dim3(H3 / 64, MBS / 64), blk, 0, stream>>>(Xh, WqkT, bqkv, QKV, MBS, H3, Hh);
  k_gemm<true><<<dim3(Hh / 64, MBS / 64), blk, 0, stream>>>(Xp, WreT, brel, REL, MBS, Hh, Hh);

  k_attn<<<dim3(Ss / 64, NHEAD, Bb), blk, 0, stream>>>(QKV, REL, rwb, rrb, CTX);

  k_gemm<false><<<dim3(Hh / 64, MBS / 64), blk, 0, stream>>>(CTX, WdnT, bdns, d_out, MBS, Hh, Hh);
}
